// ConditionalRandomField_40647570489954
// MI455X (gfx1250) — matrix operand order repaired and checked by reading, not compiled, not run on hardware
//
#include <hip/hip_runtime.h>
#include <hip/hip_bf16.h>
#include <stdint.h>

// CRF forward-algorithm NLL on gfx1250.
// Recurrence recast as exp-space GEMM so each step runs on v_wmma_f32_16x16x32_f16:
//   alpha'[j] = log( (exp(alpha-amax)) dot (exp(trans-cmax))[:,j] ) + amax + cmax[j] + emit[j]
// Workgroup = 128 threads (4 wave32), owns 16 batches; wave w owns output columns [16w,16w+16).
// E = exp(trans - cmax) is constant: its WMMA B-fragments live in VGPRs for all 1023 steps.
// Mask bytes are staged to LDS once per step (16 values) and the alpha update is a
// branchless fma blend -> no EXEC divergence in the hot loop.

typedef __attribute__((ext_vector_type(16))) _Float16 v16h;
typedef __attribute__((ext_vector_type(8)))  float    v8f;

#define B_TOT    512
#define S_LEN    1024
#define T_DIM    64
#define TB       16    // batches per workgroup (WMMA M)
#define NTHREADS 128

#define LOG2E 1.4426950408889634f
#define LN2   0.6931471805599453f

__device__ __forceinline__ float fast_exp(float x) {
#if __has_builtin(__builtin_amdgcn_exp2f)
  return __builtin_amdgcn_exp2f(x * LOG2E);   // v_exp_f32 is 2^x
#else
  return __expf(x);
#endif
}
__device__ __forceinline__ float fast_log(float x) {
#if __has_builtin(__builtin_amdgcn_logf)
  return __builtin_amdgcn_logf(x) * LN2;      // v_log_f32 is log2
#else
  return __logf(x);
#endif
}

__global__ __launch_bounds__(NTHREADS) void crf_nll_kernel(
    const float* __restrict__ feats,          // [B, S, T] f32
    const long long* __restrict__ tags,       // [B, S] i64
    const unsigned char* __restrict__ maskp,  // [B, S] bool (1 byte)
    const float* __restrict__ trans,          // [T, T] f32
    float* __restrict__ out)                  // [B] f32
{
  __shared__ float    trans_lds[T_DIM][T_DIM];  // exact f32 trans for gold path
  __shared__ _Float16 Eh[T_DIM][T_DIM];         // exp(trans - cmax) in f16
  __shared__ float    cmax[T_DIM];
  __shared__ float    alpha[TB][T_DIM];         // running log-alpha
  __shared__ _Float16 Vh[TB][T_DIM];            // exp(alpha - amax) in f16
  __shared__ float    emit_lds[TB][T_DIM];      // current emit tile
  __shared__ float    amax_lds[TB];
  __shared__ float    maskf[TB];                // per-batch mask for this step (0/1)
  __shared__ float    pred[TB][8];              // reduction scratch

  const int tid    = threadIdx.x;
  const int lane   = tid & 31;
  const int wave   = tid >> 5;
  const int wgBase = blockIdx.x * TB;

  // --- one-time setup: trans -> LDS, column max, E = exp(trans - cmax) ---
  for (int k = tid; k < T_DIM * T_DIM; k += NTHREADS)
    (&trans_lds[0][0])[k] = trans[k];
  __syncthreads();
  if (tid < T_DIM) {
    float m = trans_lds[0][tid];
    for (int i = 1; i < T_DIM; ++i) m = fmaxf(m, trans_lds[i][tid]);
    cmax[tid] = m;
  }
  __syncthreads();
  for (int k = tid; k < T_DIM * T_DIM; k += NTHREADS) {
    int i = k >> 6, j = k & 63;
    Eh[i][j] = (_Float16)fast_exp(trans_lds[i][j] - cmax[j]);
  }

  // --- init alpha = feats[:, 0, :] ---
  const int mb  = tid >> 3;         // batch row for this thread's tile work
  const int mj0 = (tid & 7) * 8;    // 8-column group
  {
    const float* src = feats + ((size_t)(wgBase + mb) * S_LEN) * T_DIM + mj0;
    #pragma unroll
    for (int q = 0; q < 8; ++q) alpha[mb][mj0 + q] = src[q];
  }
  __syncthreads();   // Eh + alpha now visible

  // --- hoist constant B fragments (32x16 f16 layout, ISA 7.12.2) ---
  v16h bf0, bf1;
  {
    int n = lane & 15, kh = lane >> 4;
    int jcol = wave * 16 + n;
    #pragma unroll
    for (int p = 0; p < 8; ++p) {
      int K0 = (p < 4 ? 2*p : 16 + 2*(p - 4)) + 8*kh;
      bf0[2*p]   = Eh[K0         ][jcol];
      bf0[2*p+1] = Eh[K0 + 1     ][jcol];
      bf1[2*p]   = Eh[K0     + 32][jcol];
      bf1[2*p+1] = Eh[K0 + 1 + 32][jcol];
    }
  }

  // --- gold accumulator: thread b (b < 16) owns batch wgBase+b ---
  float gold = 0.0f;
  int tagprev = 0;
  const long long*     tagp = tags  + (size_t)(wgBase + (tid & 15)) * S_LEN;
  const unsigned char* mkp  = maskp + (size_t)(wgBase + (tid & 15)) * S_LEN;
  if (tid < TB) {
    tagprev = (int)tagp[0];
    if (mkp[0])
      gold = feats[(size_t)(wgBase + tid) * S_LEN * T_DIM + tagprev];
  }

  // per-step incrementing pointers (avoid re-deriving addresses each step)
  const float* rowp = feats + ((size_t)(wgBase + mb) * S_LEN + 1) * T_DIM + mj0;

  for (int s = 1; s < S_LEN; ++s) {
    int tg = 0; float mkg = 0.0f;
    // phase 0: emit tile load, mask/tag staging, partial amax -- all before sync
    {
      float4 x0 = ((const float4*)rowp)[0];
      float4 x1 = ((const float4*)rowp)[1];
      *(float4*)&emit_lds[mb][mj0]     = x0;
      *(float4*)&emit_lds[mb][mj0 + 4] = x1;
      __builtin_prefetch(rowp + T_DIM, 0, 0);   // global_prefetch_b8
      rowp += T_DIM;
    }
    if (tid < TB) {
      tg  = (int)tagp[s];
      mkg = mkp[s] ? 1.0f : 0.0f;
      maskf[tid] = mkg;
    }
    {
      float m = alpha[mb][mj0];
      #pragma unroll
      for (int q = 1; q < 8; ++q) m = fmaxf(m, alpha[mb][mj0 + q]);
      pred[mb][tid & 7] = m;
    }
    __syncthreads();
    if (tid < TB) {
      float m = pred[tid][0];
      #pragma unroll
      for (int q = 1; q < 8; ++q) m = fmaxf(m, pred[tid][q]);
      amax_lds[tid] = m;
    }
    __syncthreads();
    // Vh = exp(alpha - amax) as f16
    {
      float am = amax_lds[mb];
      #pragma unroll
      for (int q = 0; q < 8; ++q)
        Vh[mb][mj0 + q] = (_Float16)fast_exp(alpha[mb][mj0 + q] - am);
    }
    __syncthreads();

    // A fragments (16x32 f16 layout, ISA 7.12.2) from Vh
    v16h a0, a1;
    {
      int m = lane & 15, hi = lane >> 4;
      #pragma unroll
      for (int p = 0; p < 8; ++p) {
        int K0 = (p < 4 ? 2*p : 16 + 2*(p - 4)) + 8*hi;
        a0[2*p]   = Vh[m][K0];
        a0[2*p+1] = Vh[m][K0 + 1];
        a1[2*p]   = Vh[m][K0 + 32];
        a1[2*p+1] = Vh[m][K0 + 33];
      }
    }
    v8f c = {};
    c = __builtin_amdgcn_wmma_f32_16x16x32_f16(false, a0, false, bf0, (short)0, c, false, false);
    c = __builtin_amdgcn_wmma_f32_16x16x32_f16(false, a1, false, bf1, (short)0, c, false, false);

    // alpha update: branchless fma blend (no EXEC divergence in hot loop)
    {
      int n = lane & 15, hi = lane >> 4;
      int j = wave * 16 + n;
      float cj = cmax[j];
      #pragma unroll
      for (int r = 0; r < 8; ++r) {
        int m = r + 8*hi;
        float msel = maskf[m];
        float aold = alpha[m][j];
        float an   = fast_log(c[r]) + amax_lds[m] + cj + emit_lds[m][j];
        alpha[m][j] = fmaf(msel, an - aold, aold);
      }
    }
    // gold path: gather emit from resident LDS tile + exact trans from LDS
    if (tid < TB) {
      gold += mkg * (emit_lds[tid][tg] + trans_lds[tagprev][tg]);
      tagprev = tg;
    }
    __syncthreads();   // protect alpha/emit_lds before next iteration
  }

  // --- norm[b] = logsumexp_j alpha[b][j]; out = norm - gold ---
  {
    float m = alpha[mb][mj0];
    #pragma unroll
    for (int q = 1; q < 8; ++q) m = fmaxf(m, alpha[mb][mj0 + q]);
    pred[mb][tid & 7] = m;
  }
  __syncthreads();
  if (tid < TB) {
    float m = pred[tid][0];
    #pragma unroll
    for (int q = 1; q < 8; ++q) m = fmaxf(m, pred[tid][q]);
    amax_lds[tid] = m;
  }
  __syncthreads();
  {
    float am = amax_lds[mb];
    float ssum = 0.f;
    #pragma unroll
    for (int q = 0; q < 8; ++q) ssum += fast_exp(alpha[mb][mj0 + q] - am);
    pred[mb][tid & 7] = ssum;
  }
  __syncthreads();
  if (tid < TB) {
    float ssum = 0.f;
    #pragma unroll
    for (int q = 0; q < 8; ++q) ssum += pred[tid][q];
    out[wgBase + tid] = (amax_lds[tid] + fast_log(ssum)) - gold;
  }
}

extern "C" void kernel_launch(void* const* d_in, const int* in_sizes, int n_in,
                              void* d_out, int out_size, void* d_ws, size_t ws_size,
                              hipStream_t stream) {
  (void)in_sizes; (void)n_in; (void)out_size; (void)d_ws; (void)ws_size;
  const float*         feats = (const float*)d_in[0];
  const long long*     tags  = (const long long*)d_in[1];
  const unsigned char* mask  = (const unsigned char*)d_in[2];
  const float*         trans = (const float*)d_in[3];
  float* out = (float*)d_out;

  dim3 grid(B_TOT / TB);     // 32 workgroups
  dim3 block(NTHREADS);      // 4 wave32 each
  hipLaunchKernelGGL(crf_nll_kernel, grid, block, 0, stream,
                     feats, tags, mask, trans, out);
}
